// NeighborAggLayer_64355789963648
// MI455X (gfx1250) — matrix operand order repaired and checked by reading, not compiled, not run on hardware
//
#include <hip/hip_runtime.h>
#include <cstdint>
#include <cstddef>

// Problem constants (from reference): B, N, K, D, H = 2, 20000, 32, 128, 64
#define BB 2
#define NN 20000
#define KK 32
#define DD 128
#define HH 64
#define MROWS (BB * NN)   // 40000
#define CBIG 256          // fused GEMM output columns: [residual | xn | self_part | P]

typedef __attribute__((ext_vector_type(16))) _Float16 v16h;
typedef __attribute__((ext_vector_type(8)))  _Float16 v8h;
typedef __attribute__((ext_vector_type(8)))  float    v8f;

// ---------------------------------------------------------------------------
// Kernel 1: fold linear chains into one 128x256 weight matrix, stored
// transposed (col-major: Wt[c*128 + k]) in f16 = the WMMA B-fragment gather
// layout. Also build the fused 256-entry f32 bias.
//   cols [0,64)    : proj_w            bias: proj_b
//   cols [64,128)  : W_nbr_w           bias: W_nbr_b
//   cols [128,192) : W_self_w @ a1_top bias: W_self_b @ a1_top
//   cols [192,256) : W_nbr_w  @ a1_bot bias: W_nbr_b  @ a1_bot
// ---------------------------------------------------------------------------
__global__ __launch_bounds__(DD) void prep_weights(
    const float* __restrict__ Wself, const float* __restrict__ Wselfb,
    const float* __restrict__ Wnbr,  const float* __restrict__ Wnbrb,
    const float* __restrict__ a1w,   const float* __restrict__ projw,
    const float* __restrict__ projb,
    _Float16* __restrict__ Wt, float* __restrict__ biasBig)
{
    int c = blockIdx.x;      // output column 0..255
    int k = threadIdx.x;     // input dim    0..127
    float v;
    if (c < HH) {
        v = projw[k * HH + c];
    } else if (c < 2 * HH) {
        v = Wnbr[k * HH + (c - HH)];
    } else if (c < 3 * HH) {
        int cc = c - 2 * HH; float s = 0.f;
        for (int j = 0; j < HH; ++j) s += Wself[k * HH + j] * a1w[j * HH + cc];
        v = s;
    } else {
        int cc = c - 3 * HH; float s = 0.f;
        for (int j = 0; j < HH; ++j) s += Wnbr[k * HH + j] * a1w[(HH + j) * HH + cc];
        v = s;
    }
    Wt[(size_t)c * DD + k] = (_Float16)v;

    if (k == 0) {
        float bv;
        if (c < HH) bv = projb[c];
        else if (c < 2 * HH) bv = Wnbrb[c - HH];
        else if (c < 3 * HH) {
            int cc = c - 2 * HH; float s = 0.f;
            for (int j = 0; j < HH; ++j) s += Wselfb[j] * a1w[j * HH + cc];
            bv = s;
        } else {
            int cc = c - 3 * HH; float s = 0.f;
            for (int j = 0; j < HH; ++j) s += Wnbrb[j] * a1w[(HH + j) * HH + cc];
            bv = s;
        }
        biasBig[c] = bv;
    }
}

// ---------------------------------------------------------------------------
// Kernel 2: x (f32) -> x16 (f16)
// ---------------------------------------------------------------------------
__global__ __launch_bounds__(256) void convert_f16(
    const float* __restrict__ x, _Float16* __restrict__ x16, int n)
{
    int i = blockIdx.x * blockDim.x + threadIdx.x;
    if (i < n) x16[i] = (_Float16)x[i];
}

// ---------------------------------------------------------------------------
// Kernel 3: fused GEMM  big = x16 (40000x128) @ WbigT + bias  via
// v_wmma_f32_16x16x32_f16. Two 16x16 tiles per wave, 4 K-steps of 32.
// A fragment (ISA 16-bit A 16x32): lanes 0-15 row M=lane take K kk+{0..7,16..23},
// lanes 16-31 take K kk+{8..15,24..31}. B fragment: lane L holds column
// tileN + L%16, same K as A: kk+8*(L/16)+{0..7,16..23} (col-major Wt).
// C/D: VGPR v -> row tileM + v + 8*(L/16), col tileN + L%16.
// ---------------------------------------------------------------------------
__global__ __launch_bounds__(256) void fused_gemm_wmma(
    const _Float16* __restrict__ x16, const _Float16* __restrict__ Wt,
    const float* __restrict__ biasBig, float* __restrict__ out)
{
    int lane = threadIdx.x & 31;
    int wave = threadIdx.x >> 5;          // 0..7, each wave does 2 N-tiles
    int tileM = blockIdx.x * 16;
    int halfSel = lane >> 4;              // 0 or 1
    int lane16 = lane & 15;

    const _Float16* arow = x16 + (size_t)(tileM + lane16) * DD;

    #pragma unroll
    for (int j = 0; j < 2; ++j) {
        int tileN = (wave * 2 + j) * 16;
        int col = tileN + lane16;
        const _Float16* bcol = Wt + (size_t)col * DD + halfSel * 8;

        v8f c = {};
        #pragma unroll
        for (int kk = 0; kk < DD; kk += 32) {
            v8h alo = *(const v8h*)(arow + kk + halfSel * 8);
            v8h ahi = *(const v8h*)(arow + kk + halfSel * 8 + 16);
            v8h blo = *(const v8h*)(bcol + kk);
            v8h bhi = *(const v8h*)(bcol + kk + 16);
            v16h a, b;
            #pragma unroll
            for (int i = 0; i < 8; ++i) {
                a[i] = alo[i]; a[i + 8] = ahi[i];
                b[i] = blo[i]; b[i + 8] = bhi[i];
            }
            c = __builtin_amdgcn_wmma_f32_16x16x32_f16(
                    false, a, false, b, (short)0, c, false, false);
        }

        float bv = biasBig[col];
        #pragma unroll
        for (int v = 0; v < 8; ++v) {
            int row = tileM + v + 8 * halfSel;
            out[(size_t)row * CBIG + col] = c[v] + bv;
        }
    }
}

// ---------------------------------------------------------------------------
// Kernel 4: per-node edge kernel. 64 threads per node (2 waves, wave32).
// Stage 32 neighbor rows of P and xn into LDS (stride 65 -> conflict-free),
// wave0 computes 32 masked attention scores + softmax via wave shuffles,
// then all 64 threads aggregate, add residual, LayerNorm, ELU.
// ---------------------------------------------------------------------------
__global__ __launch_bounds__(64) void edge_kernel(
    const float* __restrict__ big, const int* __restrict__ nidx,
    const unsigned char* __restrict__ nmask,
    const float* __restrict__ a1b, const float* __restrict__ a2w,
    const float* __restrict__ a2b,
    const float* __restrict__ lng, const float* __restrict__ lnb,
    float* __restrict__ out)
{
    __shared__ float P [KK * 65];
    __shared__ float XN[KK * 65];
    __shared__ float selfv[HH];
    __shared__ float a2s[HH];
    __shared__ float wgt[KK];
    __shared__ int   sidx[KK];
    __shared__ unsigned char smk[KK];
    __shared__ float red[4];

    int gid = blockIdx.x;                 // 0 .. MROWS-1
    int b = gid / NN, n = gid - b * NN;
    int t = threadIdx.x;                  // 0..63
    size_t rowSelf = (size_t)gid * CBIG;

    if (t < KK) {
        sidx[t] = nidx[n * KK + t];
        smk[t]  = nmask[n * KK + t];
    }
    selfv[t] = big[rowSelf + 128 + t] + a1b[t];
    a2s[t]   = a2w[t];
    __syncthreads();

    for (int k = 0; k < KK; ++k) {
        size_t r = ((size_t)b * NN + (size_t)sidx[k]) * CBIG;
        P [k * 65 + t] = big[r + 192 + t];
        XN[k * 65 + t] = big[r + 64 + t];
    }
    __syncthreads();

    if (t < KK) {   // exactly wave 0, all 32 lanes active
        float s;
        if (smk[t]) {
            s = a2b[0];
            #pragma unroll 8
            for (int h = 0; h < HH; ++h) {
                float hv = selfv[h] + P[t * 65 + h];
                hv = (hv > 0.f) ? hv : 0.2f * hv;     // leaky_relu(0.2)
                s += hv * a2s[h];
            }
        } else {
            s = -__builtin_inff();
        }
        float m = s;
        for (int off = 16; off; off >>= 1) m = fmaxf(m, __shfl_xor(m, off));
        float e = smk[t] ? __expf(s - m) : 0.f;
        float sum = e;
        for (int off = 16; off; off >>= 1) sum += __shfl_xor(sum, off);
        wgt[t] = (sum > 0.f) ? (e / sum) : 0.f;       // also covers has_nbrs==0
    }
    __syncthreads();

    float agg = 0.f;
    #pragma unroll 8
    for (int k = 0; k < KK; ++k) agg += wgt[k] * XN[k * 65 + t];
    float y = agg + big[rowSelf + t];                 // + residual

    // LayerNorm over the 64 threads (2-wave reduction)
    float s1 = y, s2 = y * y;
    for (int off = 16; off; off >>= 1) {
        s1 += __shfl_xor(s1, off);
        s2 += __shfl_xor(s2, off);
    }
    int wv = t >> 5;
    if ((t & 31) == 0) { red[wv * 2] = s1; red[wv * 2 + 1] = s2; }
    __syncthreads();
    float sum1 = red[0] + red[2];
    float sum2 = red[1] + red[3];
    float mu  = sum1 * (1.f / HH);
    float var = sum2 * (1.f / HH) - mu * mu;
    float yn = (y - mu) * rsqrtf(var + 1e-5f) * lng[t] + lnb[t];
    out[(size_t)gid * HH + t] = (yn > 0.f) ? yn : expm1f(yn);   // elu
}

// ---------------------------------------------------------------------------
// kernel_launch
// ---------------------------------------------------------------------------
extern "C" void kernel_launch(void* const* d_in, const int* in_sizes, int n_in,
                              void* d_out, int out_size, void* d_ws, size_t ws_size,
                              hipStream_t stream)
{
    const float* x      = (const float*)d_in[0];
    const int*   nidx   = (const int*)d_in[1];
    const unsigned char* nmask = (const unsigned char*)d_in[2];   // bool8
    const float* Wself  = (const float*)d_in[3];
    const float* Wselfb = (const float*)d_in[4];
    const float* Wnbr   = (const float*)d_in[5];
    const float* Wnbrb  = (const float*)d_in[6];
    const float* a1w    = (const float*)d_in[7];
    const float* a1b    = (const float*)d_in[8];
    const float* a2w    = (const float*)d_in[9];
    const float* a2b    = (const float*)d_in[10];
    const float* projw  = (const float*)d_in[11];
    const float* projb  = (const float*)d_in[12];
    const float* lng    = (const float*)d_in[13];
    const float* lnb    = (const float*)d_in[14];
    float* out = (float*)d_out;

    // workspace carve-up (all offsets 256B-aligned)
    char* ws = (char*)d_ws;
    _Float16* Wt     = (_Float16*)(ws);                 //  65,536 B
    float*    biasBig= (float*)   (ws + 65536);         //   1,024 B
    _Float16* x16    = (_Float16*)(ws + 66560);         // 10,240,000 B
    float*    big    = (float*)   (ws + 10306560);      // 40,960,000 B

    prep_weights<<<CBIG, DD, 0, stream>>>(Wself, Wselfb, Wnbr, Wnbrb,
                                          a1w, projw, projb, Wt, biasBig);

    int totalX = MROWS * DD;
    convert_f16<<<(totalX + 255) / 256, 256, 0, stream>>>(x, x16, totalX);

    fused_gemm_wmma<<<MROWS / 16, 256, 0, stream>>>(x16, Wt, biasBig, big);

    edge_kernel<<<MROWS, 64, 0, stream>>>(big, nidx, nmask,
                                          a1b, a2w, a2b, lng, lnb, out);
}
